// GNNCASimple_85341000171616
// MI455X (gfx1250) — matrix operand order repaired and checked by reading, not compiled, not run on hardware
//
#include <hip/hip_runtime.h>

typedef __attribute__((ext_vector_type(16))) _Float16 v16h;
typedef __attribute__((ext_vector_type(8)))  _Float16 v8h;
typedef __attribute__((ext_vector_type(8)))  float    v8f;

#define MT     5                  // m-tiles per block -> 80 rows (50000 = 625*80)
#define ROWS   (MT * 16)
#define LROWH  40                 // f16 LDS row stride in halves (80B, 16B-aligned)
#define LBUFH  (ROWS * LROWH)     // halves per staging buffer

// ---- CDNA5 async global->LDS copy (ASYNCcnt path) --------------------------
__device__ __forceinline__ void async_ld_b128(unsigned lds_off, unsigned long long gaddr) {
    asm volatile("global_load_async_to_lds_b128 %0, %1, off"
                 :: "v"(lds_off), "v"(gaddr) : "memory");
}
__device__ __forceinline__ void wait_async0() {
    asm volatile("s_wait_asynccnt 0x0" ::: "memory");
}
// ---- CDNA5 packed-f16 atomic add (no-return -> STOREcnt path) --------------
__device__ __forceinline__ void atomic_pk_add_f16(unsigned long long gaddr, unsigned pk) {
    asm volatile("global_atomic_pk_add_f16 %0, %1, off"
                 :: "v"(gaddr), "v"(pk) : "memory");
}

// ---------------------------------------------------------------------------
__global__ __launch_bounds__(256) void cvt_f32_to_f16(const float* __restrict__ s,
                                                      _Float16* __restrict__ d, int n) {
    int i = blockIdx.x * blockDim.x + threadIdx.x;
    if (i < n) d[i] = (_Float16)s[i];
}

// ---------------------------------------------------------------------------
// Out[M,NOUT] = act([A0|A1] @ W^T + bias)
//   A0/A1: f16 activations, W: [NOUT, K0+K1] row-major f16, acc in f32.
// Block: 80 rows x NOUT cols, 8 waves, NTPW n-tiles per wave.
// Async double-buffered LDS staging of the f16 A-tile.
// ACT: 0 relu, 1 tanh, 2 identity.  OUT16: store f16 (else f32).
// ---------------------------------------------------------------------------
template <int ACT, int NTPW, bool OUT16>
__global__ __launch_bounds__(256) void gemm_bias_act(
    const _Float16* __restrict__ A0, int K0,
    const _Float16* __restrict__ A1, int K1,
    const _Float16* __restrict__ W,
    const float* __restrict__ bias,
    void* __restrict__ OutV, int NOUT, int Nrows)
{
    __shared__ __align__(16) _Float16 At[2 * LBUFH];

    const int tid  = threadIdx.x;
    const int lane = tid & 31;
    const int wave = tid >> 5;
    const int rowbase = blockIdx.x * ROWS;
    const int K    = K0 + K1;
    const int m    = lane & 15;      // frag row (A) / col-in-tile (B/D)
    const int half = lane >> 4;      // lane group
    const int klo  = half * 16;      // A-frag K sub-offset (halves)

    const unsigned ldsbase = (unsigned)(size_t)(void*)At;  // LDS byte offset

    // accumulators pre-loaded with bias (bias depends only on column = lane)
    v8f acc[MT][NTPW];
#pragma unroll
    for (int t = 0; t < NTPW; ++t) {
        const float b = bias[(wave + t * 8) * 16 + m];
#pragma unroll
        for (int mt = 0; mt < MT; ++mt)
#pragma unroll
            for (int r = 0; r < 8; ++r) acc[mt][t][r] = b;
    }

    // stage 80x32-halves tile (kbase..kbase+31) of P into LDS buffer `buf`
    auto stage = [&](const _Float16* P, int KP, int kbase, int buf) {
        const unsigned lb = ldsbase + (unsigned)buf * (LBUFH * 2);
        for (int c = tid; c < ROWS * 4; c += 256) {      // 320 x 16B chunks
            const int r = c >> 2, q = c & 3;
            int rr = rowbase + r;
            if (rr >= Nrows) rr = Nrows - 1;             // clamp (dup row, harmless)
            const unsigned long long g = (unsigned long long)(size_t)
                (P + (size_t)rr * KP + kbase + q * 8);
            async_ld_b128(lb + (unsigned)(r * (LROWH * 2) + q * 16), g);
        }
    };

    // one 32-wide K step from LDS buffer `buf`; Wk = W + (global k offset)
    auto compute = [&](int buf, const _Float16* Wk) {
        const _Float16* Bt = At + buf * LBUFH;
        v16h bf[NTPW];
#pragma unroll
        for (int t = 0; t < NTPW; ++t)
            bf[t] = *(const v16h*)(Wk + (size_t)((wave + t * 8) * 16 + m) * K + half * 16);
#pragma unroll
        for (int mt = 0; mt < MT; ++mt) {
            const _Float16* rp = Bt + (mt * 16 + m) * LROWH;
            // A-frag: lanes 0-15 K{0..15}, lanes 16-31 K{16..31}
            const v8h alo = *(const v8h*)(rp + klo);
            const v8h ahi = *(const v8h*)(rp + klo + 8);
            const v16h af = __builtin_shufflevector(alo, ahi,
                0, 1, 2, 3, 4, 5, 6, 7, 8, 9, 10, 11, 12, 13, 14, 15);
#pragma unroll
            for (int t = 0; t < NTPW; ++t)
                acc[mt][t] = __builtin_amdgcn_wmma_f32_16x16x32_f16(
                    false, af, false, bf[t], (short)0, acc[mt][t], false, false);
        }
    };

    // -------- software-pipelined K loop over (up to) two concat phases ------
    const _Float16* Ps[2] = {A0, A1};
    const int Ksz[2] = {K0, K1};
    const int nph = (K1 > 0) ? 2 : 1;

    int buf = 0, wofs = 0;
    stage(Ps[0], Ksz[0], 0, 0);
    for (int ph = 0; ph < nph; ++ph) {
        const _Float16* P = Ps[ph];
        const int KP = Ksz[ph];
        for (int kb = 0; kb < KP; kb += 32) {
            wait_async0();        // my wave's async stores to buf complete
            __syncthreads();      // all waves: buf readable, buf^1 free
            const int nb = buf ^ 1;
            if (kb + 32 < KP)      stage(P, KP, kb + 32, nb);
            else if (ph + 1 < nph) stage(Ps[ph + 1], Ksz[ph + 1], 0, nb);
            compute(buf, W + wofs + kb);
            buf = nb;
        }
        wofs += KP;
    }

    // -------- epilogue: D row = mt*16 + 8*half + r, col = ntile*16 + m ------
    const int rb = half * 8;
#pragma unroll
    for (int t = 0; t < NTPW; ++t) {
        const int col = (wave + t * 8) * 16 + m;
#pragma unroll
        for (int mt = 0; mt < MT; ++mt) {
            const int row0 = rowbase + mt * 16 + rb;
#pragma unroll
            for (int r = 0; r < 8; ++r) {
                if (row0 + r < Nrows) {
                    float v = acc[mt][t][r];
                    if (ACT == 0)      v = fmaxf(v, 0.f);
                    else if (ACT == 1) v = tanhf(v);
                    if (OUT16)
                        ((_Float16*)OutV)[(size_t)(row0 + r) * NOUT + col] = (_Float16)v;
                    else
                        ((float*)OutV)[(size_t)(row0 + r) * NOUT + col] = v;
                }
            }
        }
    }
}

// ---------------------------------------------------------------------------
// agg[dst[e], :] += msg[src[e], :]  in f16 via packed-f16 atomics.
// 4 edges per block, 64 threads per edge, 4 halves (2 pk-atomics) per thread.
// ---------------------------------------------------------------------------
__global__ __launch_bounds__(256) void scatter_add_pk(const _Float16* __restrict__ msg,
                                                      const int* __restrict__ src,
                                                      const int* __restrict__ dst,
                                                      _Float16* __restrict__ agg,
                                                      int E, int Hf)
{
    const int sub = threadIdx.x >> 6;          // 0..3
    const int fh  = (threadIdx.x & 63) * 4;    // half index; Hf == 256
    const int e   = blockIdx.x * 4 + sub;
    if (e >= E || fh + 3 >= Hf) return;
    const int s = src[e];
    const int d = dst[e];
    const unsigned* mp = (const unsigned*)(msg + (size_t)s * Hf + fh);
    const unsigned d0 = mp[0];
    const unsigned d1 = mp[1];
    const unsigned long long o =
        (unsigned long long)(size_t)(agg + (size_t)d * Hf + fh);
    atomic_pk_add_f16(o,     d0);
    atomic_pk_add_f16(o + 4, d1);
}

// ---------------------------------------------------------------------------
extern "C" void kernel_launch(void* const* d_in, const int* in_sizes, int n_in,
                              void* d_out, int out_size, void* d_ws, size_t ws_size,
                              hipStream_t stream) {
    const float* x      = (const float*)d_in[0];
    const int*   edge   = (const int*)d_in[1];   // [2, E] int32
    const float* enc_w1 = (const float*)d_in[3];
    const float* enc_b1 = (const float*)d_in[4];
    const float* enc_w2 = (const float*)d_in[5];
    const float* enc_b2 = (const float*)d_in[6];
    const float* msg_w  = (const float*)d_in[7];
    const float* msg_b  = (const float*)d_in[8];
    const float* dec_w1 = (const float*)d_in[9];
    const float* dec_b1 = (const float*)d_in[10];
    const float* dec_w2 = (const float*)d_in[11];
    const float* dec_b2 = (const float*)d_in[12];

    const int H = in_sizes[4];             // 256
    const int D = in_sizes[12];            // 128
    const int N = in_sizes[0] / D;         // 50000
    const int E = in_sizes[1] / 2;         // 400000

    // ---- workspace partition (all f16 activations) ----
    size_t off = 0;
    auto alloc = [&](size_t bytes) -> void* {
        void* p = (char*)d_ws + off;
        off = (off + bytes + 255) & ~(size_t)255;
        return p;
    };
    _Float16* Wenc1 = (_Float16*)alloc((size_t)H * D * 2);
    _Float16* Wenc2 = (_Float16*)alloc((size_t)H * H * 2);
    _Float16* Wmsg  = (_Float16*)alloc((size_t)H * H * 2);
    _Float16* Wdec1 = (_Float16*)alloc((size_t)H * 2 * H * 2);
    _Float16* Wdec2 = (_Float16*)alloc((size_t)D * H * 2);
    _Float16* xh   = (_Float16*)alloc((size_t)N * D * 2);
    _Float16* bufA = (_Float16*)alloc((size_t)N * H * 2);  // h1 -> m -> d
    _Float16* bufB = (_Float16*)alloc((size_t)N * H * 2);  // h
    _Float16* bufC = (_Float16*)alloc((size_t)N * H * 2);  // agg
    (void)ws_size; (void)n_in; (void)out_size;

    auto cvt = [&](const float* s, _Float16* d, int n) {
        cvt_f32_to_f16<<<(n + 255) / 256, 256, 0, stream>>>(s, d, n);
    };
    cvt(enc_w1, Wenc1, H * D);
    cvt(enc_w2, Wenc2, H * H);
    cvt(msg_w,  Wmsg,  H * H);
    cvt(dec_w1, Wdec1, H * 2 * H);
    cvt(dec_w2, Wdec2, D * H);
    cvt(x,      xh,    N * D);

    const int mblocks = (N + ROWS - 1) / ROWS;   // 625

    // h1 = relu(x @ enc_w1^T + b1)                       -> bufA (f16)
    gemm_bias_act<0, 2, true><<<mblocks, 256, 0, stream>>>(
        xh, D, xh, 0, Wenc1, enc_b1, bufA, H, N);
    // h = relu(h1 @ enc_w2^T + b2)                       -> bufB (f16)
    gemm_bias_act<0, 2, true><<<mblocks, 256, 0, stream>>>(
        bufA, H, bufA, 0, Wenc2, enc_b2, bufB, H, N);
    // m = h @ msg_w^T + msg_b                            -> bufA (f16)
    gemm_bias_act<2, 2, true><<<mblocks, 256, 0, stream>>>(
        bufB, H, bufB, 0, Wmsg, msg_b, bufA, H, N);
    // agg = segment_sum(m[src], dst)                     -> bufC (f16, pk atomics)
    hipMemsetAsync(bufC, 0, (size_t)N * H * 2, stream);
    scatter_add_pk<<<(E + 3) / 4, 256, 0, stream>>>(bufA, edge, edge + E, bufC, E, H);
    // d = relu(cat(agg, h) @ dec_w1^T + b)               -> bufA (f16)
    gemm_bias_act<0, 2, true><<<mblocks, 256, 0, stream>>>(
        bufC, H, bufB, H, Wdec1, dec_b1, bufA, H, N);
    // out = tanh(d @ dec_w2^T + b)                       -> d_out (f32)
    gemm_bias_act<1, 1, false><<<mblocks, 256, 0, stream>>>(
        bufA, H, bufA, 0, Wdec2, dec_b2, d_out, D, N);
}
